// DiversityUncertainty_53833120088166
// MI455X (gfx1250) — matrix operand order repaired and checked by reading, not compiled, not run on hardware
//
#include <hip/hip_runtime.h>
#include <hip/hip_bf16.h>
#include <math.h>

typedef _Float16 half_t;
typedef __attribute__((ext_vector_type(16))) _Float16     v16h;
typedef __attribute__((ext_vector_type(8)))  float        v8f;
typedef __attribute__((ext_vector_type(4)))  unsigned int u32x4;

#define N_U    16384
#define N_L    8192
#define NZ     256
#define NC     1000
#define EPSV   1e-18f
#define NSPLIT 4
#define LCHUNK (N_L / NSPLIT)      // 2048 L columns per workgroup
#define LTILES (LCHUNK / 16)       // 128 tiles of 16
#define BSTRIDE 264                // halves per row in LDS (528B: 16B-aligned, bank-spread)

union Frag16 { v16h h; u32x4 q[2]; };
union Acc8   { v8f  v; float f[8]; };
union H4     { half_t h[4]; unsigned long long u64; };

// ---------------------------------------------------------------------------
// Kernel 1: row entropy of pred:  u[row] = -sum(exp(x)*x)
// ---------------------------------------------------------------------------
__global__ void du_entropy_kernel(const float* __restrict__ pred,
                                  float* __restrict__ uent) {
    __shared__ float s[256];
    const int row = blockIdx.x;
    const float* p = pred + (size_t)row * NC;
    float acc = 0.0f;
    for (int k = threadIdx.x; k < NC; k += 256) {
        float x = p[k];
        acc += expf(x) * x;
    }
    s[threadIdx.x] = acc;
    __syncthreads();
    for (int off = 128; off > 0; off >>= 1) {
        if (threadIdx.x < off) s[threadIdx.x] += s[threadIdx.x + off];
        __syncthreads();
    }
    if (threadIdx.x == 0) uent[row] = -s[0];
}

// ---------------------------------------------------------------------------
// Kernel 2: per-row f32 -> (f16 hi, f16 lo) split + squared L2 norm.
// One wave per row of NZ=256 elements. 8 waves per block.
// ---------------------------------------------------------------------------
__global__ void du_cvt_norm_kernel(const float* __restrict__ X,
                                   half_t* __restrict__ Hhi,
                                   half_t* __restrict__ Hlo,
                                   float* __restrict__ n2,
                                   int rows) {
    const int row  = blockIdx.x * 8 + (threadIdx.x >> 5);
    const int lane = threadIdx.x & 31;
    if (row >= rows) return;

    const float4* src = (const float4*)(X + (size_t)row * NZ);
    unsigned long long* dhi = (unsigned long long*)(Hhi + (size_t)row * NZ);
    unsigned long long* dlo = (unsigned long long*)(Hlo + (size_t)row * NZ);

    float acc = 0.0f;
#pragma unroll
    for (int j = 0; j < 2; ++j) {
        const int vi = lane * 2 + j;            // 64 float4 per row
        float4 f = src[vi];
        float vv[4] = {f.x, f.y, f.z, f.w};
        H4 hh, hl;
#pragma unroll
        for (int c = 0; c < 4; ++c) {
            float x = vv[c];
            half_t h = (half_t)x;
            hh.h[c] = h;
            hl.h[c] = (half_t)(x - (float)h);
            acc += x * x;
        }
        dhi[vi] = hh.u64;
        dlo[vi] = hl.u64;
    }
#pragma unroll
    for (int m = 16; m >= 1; m >>= 1)
        acc += __shfl_xor(acc, m, 32);
    if (lane == 0) n2[row] = acc;
}

// ---------------------------------------------------------------------------
// Kernel 3: partial min over an L-chunk of ||u - l||^2 via split-f16 WMMA.
// Block = 256 threads (8 waves). WG covers 128 U rows (16 per wave, A in
// registers); the 16-column B tile is staged in LDS once per tile and shared
// by all 8 waves, double-buffered with register prefetch.
// d2 = u2[m] + l2[n] - 2 * (u_m . l_n);  dPart[y][row] = min over chunk y.
// ---------------------------------------------------------------------------
__global__ void
__launch_bounds__(256)
du_mindist_wmma_kernel(const half_t* __restrict__ gUhi,
                       const half_t* __restrict__ gUlo,
                       const half_t* __restrict__ gLhi,
                       const half_t* __restrict__ gLlo,
                       const float* __restrict__ gU2,
                       const float* __restrict__ gL2,
                       float* __restrict__ dPart) {
    __shared__ __align__(16) half_t sBhi[2][16 * BSTRIDE];
    __shared__ __align__(16) half_t sBlo[2][16 * BSTRIDE];

    const int tid   = threadIdx.x;
    const int lane  = tid & 31;
    const int wave  = tid >> 5;
    const int Ubase = blockIdx.x * 128 + wave * 16;
    const int Lbase = blockIdx.y * LCHUNK;

    // per-lane fragment addressing constants (CDNA5 WMMA f16 layouts)
    const int m      = lane & 15;              // A row / B column within tile
    const int hiHalf = (lane >> 4) & 1;
    const int koffA  = hiHalf ? 8 : 0;
    const int koffB  = hiHalf ? 8 : 0;

    // ---- A fragments for all 8 K-chunks, loaded once, live in registers ----
    Frag16 ahi[8], alo[8];
#pragma unroll
    for (int c = 0; c < 8; ++c) {
        const half_t* ph = gUhi + (size_t)(Ubase + m) * NZ + c * 32 + koffA;
        const half_t* pl = gUlo + (size_t)(Ubase + m) * NZ + c * 32 + koffA;
        ahi[c].q[0] = *(const u32x4*)(ph);
        ahi[c].q[1] = *(const u32x4*)(ph + 16);
        alo[c].q[0] = *(const u32x4*)(pl);
        alo[c].q[1] = *(const u32x4*)(pl + 16);
    }

    float u2v[8];
#pragma unroll
    for (int r = 0; r < 8; ++r)
        u2v[r] = gU2[Ubase + r + 8 * hiHalf];

    float minv[8];
#pragma unroll
    for (int r = 0; r < 8; ++r) minv[r] = 3.0e38f;

    // B-tile staging map: thread t copies u32x4 chunks (col0, kq) & (col0+8, kq)
    const int col0 = tid >> 5;                 // 0..7
    const int kq   = tid & 31;                 // 32 u32x4 per 256-half row

    // prologue: prefetch + commit tile 0 into buffer 0
    u32x4 pfh0, pfh1, pfl0, pfl1;
    pfh0 = ((const u32x4*)(gLhi + (size_t)(Lbase + col0)     * NZ))[kq];
    pfh1 = ((const u32x4*)(gLhi + (size_t)(Lbase + col0 + 8) * NZ))[kq];
    pfl0 = ((const u32x4*)(gLlo + (size_t)(Lbase + col0)     * NZ))[kq];
    pfl1 = ((const u32x4*)(gLlo + (size_t)(Lbase + col0 + 8) * NZ))[kq];
    ((u32x4*)&sBhi[0][col0 * BSTRIDE])[kq]       = pfh0;
    ((u32x4*)&sBhi[0][(col0 + 8) * BSTRIDE])[kq] = pfh1;
    ((u32x4*)&sBlo[0][col0 * BSTRIDE])[kq]       = pfl0;
    ((u32x4*)&sBlo[0][(col0 + 8) * BSTRIDE])[kq] = pfl1;
    __syncthreads();

    for (int t = 0; t < LTILES; ++t) {
        const int lb  = Lbase + t * 16;
        const int cur = t & 1;

        // prefetch next tile into registers (overlaps with WMMA below)
        if (t + 1 < LTILES) {
            const int nb = lb + 16;
            pfh0 = ((const u32x4*)(gLhi + (size_t)(nb + col0)     * NZ))[kq];
            pfh1 = ((const u32x4*)(gLhi + (size_t)(nb + col0 + 8) * NZ))[kq];
            pfl0 = ((const u32x4*)(gLlo + (size_t)(nb + col0)     * NZ))[kq];
            pfl1 = ((const u32x4*)(gLlo + (size_t)(nb + col0 + 8) * NZ))[kq];
        }

        const float l2v = gL2[lb + m];

        Acc8 c; c.v = (v8f){0.f,0.f,0.f,0.f,0.f,0.f,0.f,0.f};
#pragma unroll
        for (int kc = 0; kc < 8; ++kc) {
            Frag16 bhi, blo;
            const half_t* bph = &sBhi[cur][m * BSTRIDE + kc * 32 + koffB];
            const half_t* bpl = &sBlo[cur][m * BSTRIDE + kc * 32 + koffB];
            bhi.q[0] = *(const u32x4*)(bph);
            bhi.q[1] = *(const u32x4*)(bph + 16);
            blo.q[0] = *(const u32x4*)(bpl);
            blo.q[1] = *(const u32x4*)(bpl + 16);

            // split-f16 product: hi*hi + hi*lo + lo*hi  (f32 accumulate)
            c.v = __builtin_amdgcn_wmma_f32_16x16x32_f16(
                false, ahi[kc].h, false, bhi.h, (short)0, c.v, false, false);
            c.v = __builtin_amdgcn_wmma_f32_16x16x32_f16(
                false, ahi[kc].h, false, blo.h, (short)0, c.v, false, false);
            c.v = __builtin_amdgcn_wmma_f32_16x16x32_f16(
                false, alo[kc].h, false, bhi.h, (short)0, c.v, false, false);
        }

#pragma unroll
        for (int r = 0; r < 8; ++r) {
            float d2 = u2v[r] + l2v - 2.0f * c.f[r];
            minv[r] = fminf(minv[r], d2);
        }

        __syncthreads();                  // everyone done reading buffer `cur`'s twin
        if (t + 1 < LTILES) {
            const int nxt = (t + 1) & 1;
            ((u32x4*)&sBhi[nxt][col0 * BSTRIDE])[kq]       = pfh0;
            ((u32x4*)&sBhi[nxt][(col0 + 8) * BSTRIDE])[kq] = pfh1;
            ((u32x4*)&sBlo[nxt][col0 * BSTRIDE])[kq]       = pfl0;
            ((u32x4*)&sBlo[nxt][(col0 + 8) * BSTRIDE])[kq] = pfl1;
        }
        __syncthreads();                  // next buffer visible to all waves
    }

    // min over the 16 lanes of each half (masks < 16 keep halves separate)
#pragma unroll
    for (int mask = 1; mask <= 8; mask <<= 1) {
#pragma unroll
        for (int r = 0; r < 8; ++r)
            minv[r] = fminf(minv[r], __shfl_xor(minv[r], mask, 32));
    }
    float* dst = dPart + (size_t)blockIdx.y * N_U + Ubase;
    if (lane == 0) {
#pragma unroll
        for (int r = 0; r < 8; ++r) dst[r] = minv[r];
    } else if (lane == 16) {
#pragma unroll
        for (int r = 0; r < 8; ++r) dst[8 + r] = minv[r];
    }
}

// ---------------------------------------------------------------------------
// Kernel 3b: combine the NSPLIT partial min-d2 -> min distance (sqrt)
// ---------------------------------------------------------------------------
__global__ void du_dfinal_kernel(const float* __restrict__ dPart,
                                 float* __restrict__ dminA) {
    const int i = blockIdx.x * 256 + threadIdx.x;
    if (i >= N_U) return;
    float dd = 3.0e38f;
#pragma unroll
    for (int y = 0; y < NSPLIT; ++y)
        dd = fminf(dd, dPart[(size_t)y * N_U + i]);
    dminA[i] = sqrtf(fmaxf(dd, 0.0f));
}

// ---------------------------------------------------------------------------
// Kernel 4: single-block min/max of uent and dmin -> scal[4]
// ---------------------------------------------------------------------------
__global__ void du_minmax_kernel(const float* __restrict__ uent,
                                 const float* __restrict__ dminA,
                                 float* __restrict__ scal) {
    __shared__ float s0[256], s1[256], s2[256], s3[256];
    float ulo = 3.0e38f, uhi = -3.0e38f, dlo = 3.0e38f, dhi = -3.0e38f;
    for (int i = threadIdx.x; i < N_U; i += 256) {
        float u = uent[i];
        ulo = fminf(ulo, u); uhi = fmaxf(uhi, u);
        float d = dminA[i];
        dlo = fminf(dlo, d); dhi = fmaxf(dhi, d);
    }
    s0[threadIdx.x] = ulo; s1[threadIdx.x] = uhi;
    s2[threadIdx.x] = dlo; s3[threadIdx.x] = dhi;
    __syncthreads();
    for (int off = 128; off > 0; off >>= 1) {
        if (threadIdx.x < off) {
            s0[threadIdx.x] = fminf(s0[threadIdx.x], s0[threadIdx.x + off]);
            s1[threadIdx.x] = fmaxf(s1[threadIdx.x], s1[threadIdx.x + off]);
            s2[threadIdx.x] = fminf(s2[threadIdx.x], s2[threadIdx.x + off]);
            s3[threadIdx.x] = fmaxf(s3[threadIdx.x], s3[threadIdx.x + off]);
        }
        __syncthreads();
    }
    if (threadIdx.x == 0) {
        scal[0] = s0[0]; scal[1] = s1[0]; scal[2] = s2[0]; scal[3] = s3[0];
    }
}

// ---------------------------------------------------------------------------
// Kernel 5: out = lambda * norm(u) + norm(d)
// norm(x) = (x - min) / ((max - min) + EPS)   (matches reference order)
// ---------------------------------------------------------------------------
__global__ void du_combine_kernel(const float* __restrict__ uent,
                                  const float* __restrict__ dminA,
                                  const float* __restrict__ scal,
                                  const int* __restrict__ lam,
                                  float* __restrict__ out) {
    const int i = blockIdx.x * 256 + threadIdx.x;
    if (i >= N_U) return;
    const float lamf = (float)(*lam);
    const float umin = scal[0], umax = scal[1];
    const float dmin = scal[2], dmax = scal[3];
    float un = (uent[i] - umin) / ((umax - umin) + EPSV);
    float dn = (dminA[i] - dmin) / ((dmax - dmin) + EPSV);
    out[i] = lamf * un + dn;
}

// ---------------------------------------------------------------------------
extern "C" void kernel_launch(void* const* d_in, const int* in_sizes, int n_in,
                              void* d_out, int out_size, void* d_ws, size_t ws_size,
                              hipStream_t stream) {
    const float* pred = (const float*)d_in[0];   // [N_U, NC]
    const float* U_z  = (const float*)d_in[1];   // [N_U, NZ]
    const float* L_z  = (const float*)d_in[2];   // [N_L, NZ]
    const int*   lam  = (const int*)d_in[3];     // scalar

    char* base = (char*)d_ws;
    size_t off = 0;
    auto carve = [&](size_t bytes) -> void* {
        void* p = base + off;
        off += (bytes + 255) & ~(size_t)255;
        return p;
    };
    float*  uent  = (float*)carve((size_t)N_U * 4);
    float*  u2    = (float*)carve((size_t)N_U * 4);
    float*  l2    = (float*)carve((size_t)N_L * 4);
    float*  dminA = (float*)carve((size_t)N_U * 4);
    float*  dPart = (float*)carve((size_t)NSPLIT * N_U * 4);
    float*  scal  = (float*)carve(4 * 4);
    half_t* Uhi   = (half_t*)carve((size_t)N_U * NZ * 2);
    half_t* Ulo   = (half_t*)carve((size_t)N_U * NZ * 2);
    half_t* Lhi   = (half_t*)carve((size_t)N_L * NZ * 2);
    half_t* Llo   = (half_t*)carve((size_t)N_L * NZ * 2);

    // 1) entropy per pred row
    du_entropy_kernel<<<N_U, 256, 0, stream>>>(pred, uent);
    // 2) split-f16 conversion + squared norms
    du_cvt_norm_kernel<<<N_U / 8, 256, 0, stream>>>(U_z, Uhi, Ulo, u2, N_U);
    du_cvt_norm_kernel<<<N_L / 8, 256, 0, stream>>>(L_z, Lhi, Llo, l2, N_L);
    // 3) WMMA partial min-distance (128 U rows per WG, L split NSPLIT ways)
    du_mindist_wmma_kernel<<<dim3(N_U / 128, NSPLIT), 256, 0, stream>>>(
        Uhi, Ulo, Lhi, Llo, u2, l2, dPart);
    du_dfinal_kernel<<<N_U / 256, 256, 0, stream>>>(dPart, dminA);
    // 4) global min/max of both vectors
    du_minmax_kernel<<<1, 256, 0, stream>>>(uent, dminA, scal);
    // 5) combine
    du_combine_kernel<<<N_U / 256, 256, 0, stream>>>(uent, dminA, scal, lam,
                                                     (float*)d_out);
}
